// ModulatedConv2d_70342974374351
// MI455X (gfx1250) — matrix operand order repaired and checked by reading, not compiled, not run on hardware
//
#include <hip/hip_runtime.h>
#include <hip/hip_bf16.h>

// ---------------- types ----------------
typedef __bf16 bf16;
typedef bf16  v16bf __attribute__((ext_vector_type(16)));
typedef bf16  v8bf  __attribute__((ext_vector_type(8)));
typedef float v8f   __attribute__((ext_vector_type(8)));

// Problem constants (from reference setup_inputs)
#define BATCH 8
#define CIN   512
#define COUT  512
#define SDIM  512
#define HH    64
#define WW    64
#define KS    3

#define CK      32                   // input channels per K-chunk
#define CHUNKS  (CIN / CK)           // 16
#define XS_U16  (6 * 66 * 32)        // x stage: 6 rows x 66 w x 32 c  (12672 u16)
#define WS_U16  (64 * 9 * 32)        // w stage: 64 oc x 9 taps x 32 c (18432 u16)
#define BUF_U16 (XS_U16 + WS_U16)    // 31104 u16 per buffer
#define XVEC    7                    // ceil(1584/256) 16B-units per thread
#define WVEC    9                    // 2304/256

__device__ __forceinline__ unsigned short f2bf(float f) {
    union { float f; unsigned u; } v; v.f = f;
    unsigned u = v.u;
    u += 0x7FFFu + ((u >> 16) & 1u);   // round-to-nearest-even
    return (unsigned short)(u >> 16);
}

__device__ __forceinline__ v16bf cat16(v8bf lo, v8bf hi) {
    return __builtin_shufflevector(lo, hi, 0,1,2,3,4,5,6,7,8,9,10,11,12,13,14,15);
}

// ---------------- 1) style: s[b][c] = 1 + affine_b[c] + coef * y[b]·affine_w[c] ----------------
__global__ void mc_style_kernel(const float* __restrict__ y,
                                const float* __restrict__ aw,
                                const float* __restrict__ ab,
                                float* __restrict__ s) {
    int idx = blockIdx.x * blockDim.x + threadIdx.x;
    if (idx >= BATCH * CIN) return;
    int b = idx / CIN, c = idx % CIN;
    const float coef = 0.044194173824159216f;            // 1/sqrt(512)
    float acc = 0.f;
    const float* yr = y + b * SDIM;
    const float* wr = aw + c * SDIM;
    for (int j = 0; j < SDIM; ++j) acc += yr[j] * wr[j];
    s[idx] = acc * coef + ab[c] + 1.0f;
}

// ---------------- 2a) wsq[o][i] = sum_kk weight^2 ----------------
__global__ void mc_wsq_kernel(const float* __restrict__ w, float* __restrict__ wsq) {
    int idx = blockIdx.x * blockDim.x + threadIdx.x;
    if (idx >= COUT * CIN) return;
    const float* p = w + idx * (KS * KS);
    float acc = 0.f;
#pragma unroll
    for (int q = 0; q < KS * KS; ++q) acc += p[q] * p[q];
    wsq[idx] = acc;
}

// ---------------- 2b) d[b][o] = rsqrt(sum_i s^2 * wsq + 1e-4) ----------------
__global__ void mc_demod_kernel(const float* __restrict__ s,
                                const float* __restrict__ wsq,
                                float* __restrict__ d) {
    int idx = blockIdx.x * blockDim.x + threadIdx.x;
    if (idx >= BATCH * COUT) return;
    int b = idx / COUT, o = idx % COUT;
    const float* sr = s + b * CIN;
    const float* wr = wsq + o * CIN;
    float acc = 1e-4f;
    for (int i = 0; i < CIN; ++i) { float sc = sr[i]; acc += sc * sc * wr[i]; }
    d[idx] = rsqrtf(acc);
}

// ---------------- 3) xmodT[b][h][w][c] = bf16(x[b][c][h][w] * s[b][c]) ----------------
__global__ void mc_xmod_kernel(const float* __restrict__ x,
                               const float* __restrict__ s,
                               unsigned short* __restrict__ xmodT) {
    __shared__ unsigned short tile[64 * 66];             // [c_l][w], padded
    int bh = blockIdx.x;                                 // BATCH*HH blocks
    int b = bh >> 6, h = bh & 63;
    int tid = threadIdx.x;                               // 256 threads
    for (int cc = 0; cc < CIN / 64; ++cc) {
#pragma unroll 4
        for (int idx = tid; idx < 64 * 64; idx += 256) { // coalesced read over w
            int c_l = idx >> 6, w = idx & 63;
            int c = cc * 64 + c_l;
            float v = x[((b * CIN + c) * HH + h) * WW + w] * s[b * CIN + c];
            tile[c_l * 66 + w] = f2bf(v);
        }
        __syncthreads();
#pragma unroll 4
        for (int idx = tid; idx < 64 * 64; idx += 256) { // coalesced write over c
            int c_l = idx & 63, w = idx >> 6;
            xmodT[((b * HH + h) * WW + w) * CIN + cc * 64 + c_l] = tile[c_l * 66 + w];
        }
        __syncthreads();
    }
}

// ---------------- 4) wbfT[o][t][i] = bf16(weight[o][i][t]) ----------------
__global__ void mc_wt_kernel(const float* __restrict__ w, unsigned short* __restrict__ wbfT) {
    int idx = blockIdx.x * blockDim.x + threadIdx.x;
    if (idx >= COUT * 9 * CIN) return;
    int i = idx % CIN;
    int t = (idx / CIN) % 9;
    int o = idx / (9 * CIN);
    wbfT[idx] = f2bf(w[(o * CIN + i) * 9 + t]);
}

// ---------------- 5) WMMA implicit-GEMM conv ----------------
// grid: (COUT/64, HH/4, BATCH), block: 256 threads (8 waves)
// block tile: M=64 oc x N=256 px (4 rows x 64 w); each wave: 4 M-subtiles x 2 N-subtiles
// double-buffered LDS (dynamic, 2 x 62208 B) with register prefetch of the next K-chunk
extern __shared__ unsigned short mc_smem[];

__global__ void __launch_bounds__(256)
mc_conv_wmma_kernel(const unsigned short* __restrict__ xmodT,
                    const unsigned short* __restrict__ wbfT,
                    const float* __restrict__ dmod,
                    const float* __restrict__ bias,
                    float* __restrict__ out) {
    const int o0 = blockIdx.x * 64;
    const int h0 = blockIdx.y * 4;
    const int b  = blockIdx.z;
    const int tid = threadIdx.x;
    const int lane = tid & 31;                 // wave32
    const int wid = tid >> 5;
    const int lane15 = lane & 15, laneHi = lane >> 4;
    const int nj0 = wid * 2;                   // this wave's 2 N-subtiles

    v8f acc[4][2];
#pragma unroll
    for (int mi = 0; mi < 4; ++mi) { acc[mi][0] = (v8f)(0.0f); acc[mi][1] = (v8f)(0.0f); }

    uint4 xg[XVEC], wg[WVEC];

    auto loadChunk = [&](int ci) {
#pragma unroll
        for (int k = 0; k < XVEC; ++k) {       // 6 rows x 66 w x 32 c, zero halo
            int idx = tid + k * 256;
            uint4 v = make_uint4(0, 0, 0, 0);
            if (idx < 6 * 66 * 4) {
                int r = idx / 264, rem = idx % 264;
                int w1 = rem >> 2, q = rem & 3;
                int hh = h0 - 1 + r, ww = w1 - 1;
                if (hh >= 0 && hh < HH && ww >= 0 && ww < WW)
                    v = *(const uint4*)(xmodT + (((b * HH + hh) * WW + ww) * CIN + ci * CK + q * 8));
            }
            xg[k] = v;
        }
#pragma unroll
        for (int k = 0; k < WVEC; ++k) {       // 64 oc x 9 taps x 32 c
            int idx = tid + k * 256;
            int o_l = idx / 36, rem = idx % 36;
            int t = rem >> 2, q = rem & 3;
            wg[k] = *(const uint4*)(wbfT + (((o0 + o_l) * 9 + t) * CIN + ci * CK + q * 8));
        }
    };

    auto storeChunk = [&](int buf) {           // LDS linear index == idx*8 by construction
        unsigned short* base = mc_smem + buf * BUF_U16;
#pragma unroll
        for (int k = 0; k < XVEC; ++k) {
            int idx = tid + k * 256;
            if (idx < 6 * 66 * 4) *(uint4*)(base + idx * 8) = xg[k];
        }
        unsigned short* wbase = base + XS_U16;
#pragma unroll
        for (int k = 0; k < WVEC; ++k) {
            int idx = tid + k * 256;
            *(uint4*)(wbase + idx * 8) = wg[k];
        }
    };

    auto computeChunk = [&](int buf) {
        const unsigned short* xsb  = mc_smem + buf * BUF_U16;
        const unsigned short* wshb = xsb + XS_U16;
        // per-lane bases; tap offsets become compile-time ds-offset immediates
        const int p0 = nj0 * 16 + lane15;      // pixel for N-subtile 0
        const int p1 = p0 + 16;
        const unsigned short* bBase0 = xsb + (((p0 >> 6) * 66) + (p0 & 63)) * 32 + laneHi * 16;
        const unsigned short* bBase1 = xsb + (((p1 >> 6) * 66) + (p1 & 63)) * 32 + laneHi * 16;
        const unsigned short* aBase  = wshb + lane15 * (9 * 32) + laneHi * 16;
#pragma unroll
        for (int kh = 0; kh < 3; ++kh) {
#pragma unroll
            for (int kw = 0; kw < 3; ++kw) {
                const int t = kh * 3 + kw;
                const int xoff = (kh * 66 + kw) * 32;
                // B frag (32x16): col = lane15, k = laneHi*16 + e (contiguous 32B)
                v16bf b0 = cat16(*(const v8bf*)(bBase0 + xoff),
                                 *(const v8bf*)(bBase0 + xoff + 8));
                v16bf b1 = cat16(*(const v8bf*)(bBase1 + xoff),
                                 *(const v8bf*)(bBase1 + xoff + 8));
#pragma unroll
                for (int mi = 0; mi < 4; ++mi) {
                    // A frag (16x32): row = lane15, k = laneHi*16 + e (contiguous 32B)
                    const unsigned short* ap = aBase + mi * (16 * 9 * 32) + t * 32;
                    v16bf a = cat16(*(const v8bf*)ap, *(const v8bf*)(ap + 8));
                    acc[mi][0] = __builtin_amdgcn_wmma_f32_16x16x32_bf16(
                        false, a, false, b0, (short)0, acc[mi][0], false, false);
                    acc[mi][1] = __builtin_amdgcn_wmma_f32_16x16x32_bf16(
                        false, a, false, b1, (short)0, acc[mi][1], false, false);
                }
            }
        }
    };

    // software pipeline: prefetch chunk ci+1 while computing chunk ci
    loadChunk(0);
    storeChunk(0);
    __syncthreads();
    for (int ci = 0; ci < CHUNKS; ++ci) {
        if (ci + 1 < CHUNKS) loadChunk(ci + 1);
        computeChunk(ci & 1);
        if (ci + 1 < CHUNKS) storeChunk((ci + 1) & 1);
        __syncthreads();
    }

    // epilogue: demod * acc + bias ; C layout: VGPR jj -> M = jj + laneHi*8, N = lane15
#pragma unroll
    for (int jn = 0; jn < 2; ++jn) {
        int p = (nj0 + jn) * 16 + lane15;
        int h = h0 + (p >> 6), w = p & 63;
#pragma unroll
        for (int mi = 0; mi < 4; ++mi) {
#pragma unroll
            for (int jj = 0; jj < 8; ++jj) {
                int o = o0 + mi * 16 + laneHi * 8 + jj;
                out[((b * COUT + o) * HH + h) * WW + w] =
                    acc[mi][jn][jj] * dmod[b * COUT + o] + bias[o];
            }
        }
    }
}

// ---------------- launcher ----------------
extern "C" void kernel_launch(void* const* d_in, const int* in_sizes, int n_in,
                              void* d_out, int out_size, void* d_ws, size_t ws_size,
                              hipStream_t stream) {
    const float* x   = (const float*)d_in[0];
    const float* y   = (const float*)d_in[1];
    const float* aw  = (const float*)d_in[2];
    const float* ab  = (const float*)d_in[3];
    const float* wgt = (const float*)d_in[4];
    const float* bia = (const float*)d_in[5];
    float* out = (float*)d_out;

    char* ws = (char*)d_ws;
    size_t off = 0;
    float* s    = (float*)(ws + off); off += (size_t)BATCH * CIN * 4;
    float* dmd  = (float*)(ws + off); off += (size_t)BATCH * COUT * 4;
    float* wsq  = (float*)(ws + off); off += (size_t)COUT * CIN * 4;
    unsigned short* xmodT = (unsigned short*)(ws + off);
    off += (size_t)BATCH * HH * WW * CIN * 2;
    unsigned short* wbfT = (unsigned short*)(ws + off);
    off += (size_t)COUT * 9 * CIN * 2;
    (void)ws_size; (void)in_sizes; (void)n_in; (void)out_size;

    mc_style_kernel<<<(BATCH * CIN + 255) / 256, 256, 0, stream>>>(y, aw, ab, s);
    mc_wsq_kernel<<<(COUT * CIN + 255) / 256, 256, 0, stream>>>(wgt, wsq);
    mc_demod_kernel<<<(BATCH * COUT + 255) / 256, 256, 0, stream>>>(s, wsq, dmd);
    mc_xmod_kernel<<<BATCH * HH, 256, 0, stream>>>(x, s, xmodT);
    mc_wt_kernel<<<(COUT * 9 * CIN + 255) / 256, 256, 0, stream>>>(wgt, wbfT);

    dim3 grid(COUT / 64, HH / 4, BATCH);
    size_t ldsBytes = (size_t)2 * BUF_U16 * sizeof(unsigned short);  // 124416 B
    mc_conv_wmma_kernel<<<grid, 256, ldsBytes, stream>>>(xmodT, wbfT, dmd, bia, out);
}
